// RankAwareContrastiveLoss_22995254903115
// MI455X (gfx1250) — compile-verified, matrix operand order checked by reading, not run on hardware
//
#include <hip/hip_runtime.h>

#define B_N 4096
#define D_K 128
#define INV_TEMP (1.0f / 0.07f)
#define INV_MARGIN (1.0f / 0.15f)
#define LSTR 136   // padded LDS row stride in halves (256B row + 16B pad = 272B)

typedef _Float16 half16 __attribute__((ext_vector_type(16)));
typedef _Float16 half8  __attribute__((ext_vector_type(8)));
typedef float    f32x8  __attribute__((ext_vector_type(8)));
typedef unsigned int u32x4 __attribute__((ext_vector_type(4)));
typedef int          i32x8 __attribute__((ext_vector_type(8)));
typedef int          i32x4 __attribute__((ext_vector_type(4)));

// Monotonic unsigned key: u(a) < u(b)  <=>  a < b  (all finite floats)
__device__ __forceinline__ unsigned int fkey(float f) {
    unsigned int b = __float_as_uint(f);
    return b ^ ((unsigned int)((int)b >> 31) | 0x80000000u);
}

// ---- Tensor Data Mover: load a 64-row x 128-half tile (row stride 128 halves
// in memory) into LDS with 16B padding after every 256B row (-> LSTR layout).
__device__ __forceinline__ void tdm_load_tile(unsigned int lds_addr,
                                              const _Float16* gptr) {
    const unsigned long long ga = (unsigned long long)(size_t)gptr;
    u32x4 g0;
    g0[0] = 1u;                                            // count=1, user descriptor
    g0[1] = lds_addr;                                      // lds_addr (bytes)
    g0[2] = (unsigned int)(ga & 0xFFFFFFFFull);            // global_addr[31:0]
    g0[3] = (unsigned int)((ga >> 32) & 0x01FFFFFFull)     // global_addr[56:32]
            | (2u << 30);                                  // type=2 ("image")
    i32x8 g1;
    g1[0] = (int)((1u << 16)       // data_size = 1 -> 2 bytes
                | (1u << 20)       // pad_enable
                | (5u << 22)       // pad_interval code 5 -> every 64 DWORDs (256B)
                | (3u << 25));     // pad_amount  code 3 -> 4 DWORDs (16B)
    g1[1] = (int)((unsigned)D_K << 16);   // tensor_dim0 = 128 (bits [79:48], low16)
    g1[2] = (int)((unsigned)B_N << 16);   // tensor_dim0 hi16=0 | tensor_dim1 = 4096 (low16)
    g1[3] = (int)((unsigned)D_K << 16);   // tensor_dim1 hi16=0 | tile_dim0 = 128
    g1[4] = 64;                           // tile_dim1 = 64, tile_dim2 = 0
    g1[5] = D_K;                          // tensor_dim0_stride = 128 (low 32)
    g1[6] = 0;                            // stride0 hi16 = 0 | stride1 lo16 = 0
    g1[7] = 0;                            // stride1 hi32 = 0
    const i32x4 z4 = {0, 0, 0, 0};        // groups 2/3 unused (2D tensor)
    const i32x8 z8 = {0, 0, 0, 0, 0, 0, 0, 0};
    // amdgpu-toolchain (clang-23) form: 6 args, extra int32x8 group before cpol
    __builtin_amdgcn_tensor_load_to_lds(g0, g1, z4, z4, z8, 0);
}

// ---------------------------------------------------------------- init ----
__global__ void init_scalars(unsigned int* cnt) {
    *cnt = 0u;
}

// ----------------------------------------------------- normalize (3 mats) -
// grid (4096, 3), block 128.  branch 0: drug, 1: se, 2: drug*se
__global__ void normalize3(const float* __restrict__ drug,
                           const float* __restrict__ se,
                           _Float16* __restrict__ z) {
    __shared__ float red[128];
    const int row = blockIdx.x, br = blockIdx.y, e = threadIdx.x;
    const float a = drug[(size_t)row * D_K + e];
    const float b = se[(size_t)row * D_K + e];
    const float v = (br == 0) ? a : ((br == 1) ? b : a * b);
    red[e] = v * v;
    __syncthreads();
    for (int s = 64; s > 0; s >>= 1) {
        if (e < s) red[e] += red[e + s];
        __syncthreads();
    }
    const float norm = fmaxf(sqrtf(red[0]), 1e-12f);
    z[((size_t)br * B_N + row) * D_K + e] = (_Float16)(v / norm);
}

// ------------------------------------------------- global negative count --
__global__ void count_negs(const float* __restrict__ labels, unsigned int* cnt) {
    __shared__ unsigned int red[256];
    const int i = blockIdx.x;
    const float li = labels[i];
    unsigned int c = 0;
    for (int j = threadIdx.x; j < B_N; j += 256)
        c += (unsigned int)((j != i) && (fabsf(li - labels[j]) > 1.0f));
    red[threadIdx.x] = c;
    __syncthreads();
    for (int s = 128; s > 0; s >>= 1) {
        if (threadIdx.x < s) red[threadIdx.x] += red[threadIdx.x + s];
        __syncthreads();
    }
    if (threadIdx.x == 0) atomicAdd(cnt, red[0]);
}

__global__ void compute_k(const unsigned int* cnt, int* kv, unsigned int* anyneg) {
    const float mean = (float)(*cnt) / (float)B_N;
    int k = (int)(mean * 0.5f);         // truncation, matches astype(int32)
    k = k < 1 ? 1 : k;
    k = k > (B_N - 1) ? (B_N - 1) : k;
    *kv = k;
    *anyneg = (*cnt) > 0u ? 1u : 0u;
}

// ------------------------------------------------------- sim = z z^T / T --
// grid (64, 64) of 64x64 macro-tiles, block 256 (8 waves).
// A/B tiles staged by the Tensor Data Mover (TENSORcnt), which also applies
// the bank-spreading LDS padding in hardware. Wave w -> tile row (w>>1),
// tile cols {2*(w&1), 2*(w&1)+1} so the A fragment is reused across 2 WMMAs.
// Only upper triangle (bx >= by) computed; mirrored stores exploit symmetry.
__global__ void sim_wmma(const _Float16* __restrict__ z, float* __restrict__ sim) {
    __shared__ __align__(16) _Float16 As[64 * LSTR];
    __shared__ __align__(16) _Float16 Bs[64 * LSTR];

    const int bx = blockIdx.x, by = blockIdx.y;
    if (bx < by) return;  // symmetry: skip lower-triangular blocks
    const int i0 = by * 64, j0 = bx * 64;
    const int tid = threadIdx.x;
    const int lane = tid & 31, wave = tid >> 5;

    // One TDM descriptor per tile; issued by lane 0 of waves 0 and 1
    // (TDM ignores EXEC; TENSORcnt is per-wave, every wave's wait is correct).
    if (tid == 0)
        tdm_load_tile((unsigned int)(size_t)&As[0], &z[(size_t)i0 * D_K]);
    if (tid == 32)
        tdm_load_tile((unsigned int)(size_t)&Bs[0], &z[(size_t)j0 * D_K]);
    __builtin_amdgcn_s_wait_tensorcnt(0);
    __syncthreads();

    const int m = lane & 15, h = lane >> 4;  // 16-bit A-layout: lanes 16-31 get K+8

    const int tr  = wave >> 1;
    const int tc0 = (wave & 1) * 2;
    f32x8 c0 = {}, c1 = {};
    const int arow  = (tr * 16 + m) * LSTR;
    const int brow0 = (tc0 * 16 + m) * LSTR;
    const int brow1 = brow0 + 16 * LSTR;

#pragma unroll
    for (int kb = 0; kb < 4; ++kb) {
        const int base = kb * 32 + h * 8;
        half8 alo = *(const half8*)&As[arow + base];
        half8 ahi = *(const half8*)&As[arow + base + 16];
        half16 a = __builtin_shufflevector(alo, ahi, 0,1,2,3,4,5,6,7,8,9,10,11,12,13,14,15);

        half8 b0lo = *(const half8*)&Bs[brow0 + base];
        half8 b0hi = *(const half8*)&Bs[brow0 + base + 16];
        half16 b0 = __builtin_shufflevector(b0lo, b0hi, 0,1,2,3,4,5,6,7,8,9,10,11,12,13,14,15);
        c0 = __builtin_amdgcn_wmma_f32_16x16x32_f16(false, a, false, b0, (short)0, c0,
                                                    false, false);

        half8 b1lo = *(const half8*)&Bs[brow1 + base];
        half8 b1hi = *(const half8*)&Bs[brow1 + base + 16];
        half16 b1 = __builtin_shufflevector(b1lo, b1hi, 0,1,2,3,4,5,6,7,8,9,10,11,12,13,14,15);
        c1 = __builtin_amdgcn_wmma_f32_16x16x32_f16(false, a, false, b1, (short)0, c1,
                                                    false, false);
    }

    // C/D layout: VGPR r, lane -> (M = r + 8h, N = m)
    const int gr0 = i0 + tr * 16 + h * 8;
    const int gcA = j0 + tc0 * 16 + m;
    const int gcB = gcA + 16;
#pragma unroll
    for (int r = 0; r < 8; ++r) {
        const int gr = gr0 + r;
        const float v0 = c0[r] * INV_TEMP;
        const float v1 = c1[r] * INV_TEMP;
        sim[(size_t)gr * B_N + gcA] = v0;
        sim[(size_t)gcA * B_N + gr] = v0;  // symmetric mirror (bitwise identical dot)
        sim[(size_t)gr * B_N + gcB] = v1;
        sim[(size_t)gcB * B_N + gr] = v1;
    }
}

// --------------------------------------------- per-row select + loss -------
// One block (256 threads) per row. Row sim + labels live fully in LDS.
__global__ void row_loss(const float* __restrict__ sim,
                         const float* __restrict__ labels,
                         const int* __restrict__ kv,
                         const unsigned int* __restrict__ anyneg_p,
                         float* __restrict__ rowloss) {
    __shared__ float srow[B_N];            // 16 KB
    __shared__ float slab[B_N];            // 16 KB
    __shared__ unsigned int hist[256];
    __shared__ float redE[256];
    __shared__ float redS1[256];
    __shared__ float redS2[256];
    __shared__ unsigned int sh_prefix;
    __shared__ int sh_kr;

    const int i = blockIdx.x, tid = threadIdx.x;
    for (int j = tid; j < B_N; j += 256) {
        srow[j] = sim[(size_t)i * B_N + j];
        slab[j] = labels[j];
    }
    __syncthreads();
    const float li = slab[i];

    // Row max + this row's negative count
    float lmax = -INFINITY;
    unsigned int lcnt = 0;
    for (int j = tid; j < B_N; j += 256) {
        lmax = fmaxf(lmax, srow[j]);
        lcnt += (unsigned int)((j != i) && (fabsf(li - slab[j]) > 1.0f));
    }
    redE[tid] = lmax;
    hist[tid] = lcnt;
    __syncthreads();
    for (int s = 128; s > 0; s >>= 1) {
        if (tid < s) {
            redE[tid] = fmaxf(redE[tid], redE[tid + s]);
            hist[tid] += hist[tid + s];
        }
        __syncthreads();
    }
    const float rmax = redE[0];
    const int cneg = (int)hist[0];
    const int k = *kv;
    const unsigned int anyneg = *anyneg_p;
    __syncthreads();

    // Exact k-th largest negative sim via 4-pass byte radix select on fkey.
    unsigned int T = 0u;  // key threshold; 0 => accept all negatives
    if (anyneg && cneg > k) {
        if (tid == 0) { sh_prefix = 0u; sh_kr = k; }
        __syncthreads();
        for (int pass = 0; pass < 4; ++pass) {
            const int shift = 24 - 8 * pass;
            hist[tid] = 0u;
            __syncthreads();
            const unsigned int prefix = sh_prefix;
            const unsigned int himask = (pass == 0) ? 0u : (0xFFFFFFFFu << (shift + 8));
            for (int j = tid; j < B_N; j += 256) {
                if ((j != i) && (fabsf(li - slab[j]) > 1.0f)) {
                    const unsigned int u = fkey(srow[j]);
                    if ((u & himask) == prefix)
                        atomicAdd(&hist[(u >> shift) & 255u], 1u);
                }
            }
            __syncthreads();
            if (tid == 0) {
                int kr = sh_kr, bsel = 0;
                for (int b = 255; b >= 0; --b) {
                    const int cb = (int)hist[b];
                    if (kr <= cb) { bsel = b; break; }
                    kr -= cb;
                }
                sh_kr = kr;
                sh_prefix = prefix | ((unsigned int)bsel << shift);
            }
            __syncthreads();
        }
        T = sh_prefix;
    }
    __syncthreads();

    // Final sums: E = sum exp(logit) over denom mask, S1 = sum pw, S2 = sum pw*logit
    float E = 0.0f, S1 = 0.0f, S2 = 0.0f;
    for (int j = tid; j < B_N; j += 256) {
        const float lj = slab[j];
        const float ld = fabsf(li - lj);
        const bool selfok = (j != i);
        const float pw = selfok ? __expf(-ld * INV_MARGIN) : 0.0f;
        const bool neg = selfok && (ld > 1.0f);
        const float sv = srow[j];
        const float lg = sv - rmax;
        const bool dm = anyneg ? ((neg && (fkey(sv) >= T)) || (pw > 0.1f)) : selfok;
        if (dm) E += __expf(lg);
        S1 += pw;
        S2 += pw * lg;
    }
    redE[tid] = E; redS1[tid] = S1; redS2[tid] = S2;
    __syncthreads();
    for (int s = 128; s > 0; s >>= 1) {
        if (tid < s) {
            redE[tid]  += redE[tid + s];
            redS1[tid] += redS1[tid + s];
            redS2[tid] += redS2[tid + s];
        }
        __syncthreads();
    }
    if (tid == 0) {
        const float Et = redE[0] + 1e-8f;
        const float S1t = redS1[0];
        const float S2t = redS2[0];
        // sum pw*(logit - log(E)) = S2 - log(E)*S1
        rowloss[i] = -(S2t - logf(Et) * S1t) / fmaxf(S1t, 1e-8f);
    }
}

// ---------------------------------------------------------------- finalize -
__global__ void finalize(const float* __restrict__ rowloss, float* __restrict__ out) {
    __shared__ float red[256];
    float s = 0.0f;
    for (int j = threadIdx.x; j < 3 * B_N; j += 256) s += rowloss[j];
    red[threadIdx.x] = s;
    __syncthreads();
    for (int st = 128; st > 0; st >>= 1) {
        if (threadIdx.x < st) red[threadIdx.x] += red[threadIdx.x + st];
        __syncthreads();
    }
    if (threadIdx.x == 0) out[0] = red[0] / (3.0f * (float)B_N);
}

// ---------------------------------------------------------------- launch ---
extern "C" void kernel_launch(void* const* d_in, const int* in_sizes, int n_in,
                              void* d_out, int out_size, void* d_ws, size_t ws_size,
                              hipStream_t stream) {
    (void)in_sizes; (void)n_in; (void)out_size; (void)ws_size;
    const float* drug   = (const float*)d_in[0];
    const float* se     = (const float*)d_in[1];
    const float* labels = (const float*)d_in[2];

    char* ws = (char*)d_ws;
    float*    sim     = (float*)ws;                                     // 64 MB (reused per branch)
    _Float16* z       = (_Float16*)(ws + (size_t)B_N * B_N * 4);        // 3 MB (3 branches)
    float*    rowloss = (float*)(ws + (size_t)B_N * B_N * 4 + (size_t)3 * B_N * D_K * 2);
    unsigned int* cnt    = (unsigned int*)((char*)rowloss + (size_t)3 * B_N * 4);
    int*          kv     = (int*)(cnt + 1);
    unsigned int* anyneg = (unsigned int*)(kv + 1);

    init_scalars<<<1, 1, 0, stream>>>(cnt);
    normalize3<<<dim3(B_N, 3), 128, 0, stream>>>(drug, se, z);
    count_negs<<<B_N, 256, 0, stream>>>(labels, cnt);
    compute_k<<<1, 1, 0, stream>>>(cnt, kv, anyneg);

    for (int br = 0; br < 3; ++br) {
        sim_wmma<<<dim3(64, 64), 256, 0, stream>>>(z + (size_t)br * B_N * D_K, sim);
        row_loss<<<B_N, 256, 0, stream>>>(sim, labels, kv, anyneg, rowloss + (size_t)br * B_N);
    }
    finalize<<<1, 256, 0, stream>>>(rowloss, (float*)d_out);
}
